// DeformableConv2D_19370302505415
// MI455X (gfx1250) — matrix operand order repaired and checked by reading, not compiled, not run on hardware
//
#include <hip/hip_runtime.h>

// ---- problem constants (B,H,W,C,F) = (16,128,128,256,256) ----
#define BB   16
#define HH   128
#define WW   128
#define CC   256
#define FF   256
#define HWPX (HH * WW)          // 16384
#define NPIX (BB * HWPX)        // 262144
#define KTOT (9 * CC)           // 2304

// GEMM tiling: block 64(M) x 128(N) x 32(K); 8 waves as 2(M) x 4(N),
// each wave owns a 32x32 output tile = 4 x v_wmma_f32_16x16x32_bf16 per K-step.
#define TILE_M 64
#define TILE_N 128
#define TILE_K 32
#define NKT    (KTOT / TILE_K)  // 72
#define LDK    (TILE_K + 8)     // padded LDS row (bf16 elems)

typedef __attribute__((ext_vector_type(16))) __bf16 v16bf;
typedef __attribute__((ext_vector_type(8)))  __bf16 v8bf;
typedef __attribute__((ext_vector_type(4)))  __bf16 v4bf;
typedef __attribute__((ext_vector_type(8)))  float  v8f;
typedef __attribute__((ext_vector_type(4)))  int    v4i;

#define AS1 __attribute__((address_space(1)))
#define AS3 __attribute__((address_space(3)))

#if defined(__has_builtin)
#if __has_builtin(__builtin_amdgcn_global_load_async_to_lds_b128) && \
    __has_builtin(__builtin_amdgcn_s_wait_asynccnt)
#define USE_ASYNC 1
#endif
#endif

// -------------------------------------------------------------------------
// K1: offset conv (3x3xCx2, SAME) -> integer sampling index lin[p] = y*W + x
// -------------------------------------------------------------------------
__global__ __launch_bounds__(256) void offsets_lin_kernel(
    const float* __restrict__ in, const float* __restrict__ w_off,
    const float* __restrict__ b_off, int* __restrict__ lin)
{
    __shared__ float sw[9 * CC * 2];              // 4608 floats = 18 KB
    for (int i = threadIdx.x; i < 9 * CC * 2; i += 256) sw[i] = w_off[i];
    __syncthreads();

    const int p = blockIdx.x * 256 + threadIdx.x; // one pixel per thread
    const int b = p >> 14, rem = p & (HWPX - 1);
    const int h = rem >> 7, w = rem & (WW - 1);

    float a0 = b_off[0], a1 = b_off[1];
    #pragma unroll
    for (int tap = 0; tap < 9; ++tap) {
        const int hh = h + tap / 3 - 1, ww = w + tap % 3 - 1;
        if (hh < 0 || hh >= HH || ww < 0 || ww >= WW) continue;
        const float* src = in + ((size_t)(b << 14) + hh * WW + ww) * CC;
        const float* wt  = &sw[tap * CC * 2];
        for (int c = 0; c < CC; c += 4) {
            float4 v = *(const float4*)(src + c);
            a0 += v.x * wt[c * 2 + 0] + v.y * wt[c * 2 + 2] +
                  v.z * wt[c * 2 + 4] + v.w * wt[c * 2 + 6];
            a1 += v.x * wt[c * 2 + 1] + v.y * wt[c * 2 + 3] +
                  v.z * wt[c * 2 + 5] + v.w * wt[c * 2 + 7];
        }
    }
    const float yf = fminf(fmaxf((float)h + a0, 0.f), (float)(HH - 1));
    const float xf = fminf(fmaxf((float)w + a1, 0.f), (float)(WW - 1));
    lin[p] = ((int)yf) * WW + (int)xf;
}

// -------------------------------------------------------------------------
// K2: inputs f32 -> bf16
// -------------------------------------------------------------------------
__global__ __launch_bounds__(256) void cvt_inputs_kernel(
    const float* __restrict__ in, __bf16* __restrict__ out)
{
    const size_t i = ((size_t)blockIdx.x * 256 + threadIdx.x) * 4;
    float4 v = *(const float4*)(in + i);
    v4bf o = { (__bf16)v.x, (__bf16)v.y, (__bf16)v.z, (__bf16)v.w };
    *(v4bf*)(out + i) = o;
}

// -------------------------------------------------------------------------
// K3: w_def (3,3,C,F) -> wB[f][tap*C+c] bf16; also zero-fill the pad row
// -------------------------------------------------------------------------
__global__ __launch_bounds__(256) void cvt_wdef_kernel(
    const float* __restrict__ w, __bf16* __restrict__ out,
    __bf16* __restrict__ zpad)
{
    const int o = blockIdx.x * 256 + threadIdx.x;   // < F * KTOT
    const int f = o / KTOT, r = o % KTOT;
    out[o] = (__bf16)w[(size_t)r * FF + f];
    if (blockIdx.x == 0 && threadIdx.x < 64) zpad[threadIdx.x] = (__bf16)0.f;
}

// -------------------------------------------------------------------------
// K4: implicit-GEMM deformable conv (gather fused into A loads)
// -------------------------------------------------------------------------
__global__ __launch_bounds__(256) void dconv_gemm_kernel(
    const __bf16* __restrict__ inB,   // [NPIX][C] bf16
    const __bf16* __restrict__ wB,    // [F][KTOT] bf16
    const float*  __restrict__ bias,  // [F]
    const int*    __restrict__ lin,   // [NPIX]
    const __bf16* __restrict__ zpad,  // zero pad for OOB gather rows
    float* __restrict__ out)          // [NPIX][F]
{
#if USE_ASYNC
    __shared__ __bf16 At[3][TILE_M][LDK];
    __shared__ __bf16 Bt[3][TILE_N][LDK];
#else
    __shared__ __bf16 At[1][TILE_M][LDK];
    __shared__ __bf16 Bt[1][TILE_N][LDK];
#endif

    const int tid   = threadIdx.x;
    const int lane  = tid & 31;
    const int wave  = tid >> 5;
    const int waveM = wave & 1;       // 0..1
    const int waveN = wave >> 1;      // 0..3
    const int pbase = blockIdx.x * TILE_M;
    const int nbase = blockIdx.y * TILE_N;

    // A loader: 64 rows x 4 groups of 8 bf16 (1 b128 / thread / stage)
    const int arow = tid >> 2;
    const int acg  = (tid & 3) << 3;
    const int p    = pbase + arow;
    const int bimg = p >> 14;
    const int prem = p & (HWPX - 1);
    const int ph   = prem >> 7;
    const int pw   = prem & (WW - 1);
    const size_t ibase = (size_t)bimg << 14;

    // B loader: 128 rows(f) x 2 groups of 16 bf16 (2 b128 / thread / stage)
    const int bfr = tid >> 1;
    const int bkg = (tid & 1) << 4;
    const __bf16* wrow = wB + (size_t)(nbase + bfr) * KTOT + bkg;

    const int half = lane >> 4;
    const int l16  = lane & 15;
    const int am0  = (waveM << 5) + l16;         // A rows for sub-tiles 0/1
    const int bn0  = (waveN << 5) + l16;         // B rows (f) for sub-tiles 0/1

    v8f acc00 = {}, acc01 = {}, acc10 = {}, acc11 = {};

#if USE_ASYNC
    // ---- stage issue: 3 async b128 instructions per wave per stage ----
    #define ISSUE_STAGE(buf, kt)                                              \
    do {                                                                      \
        const int k_  = (kt) * TILE_K;                                        \
        const int tap = k_ >> 8;                                              \
        const int cb  = k_ & (CC - 1);                                        \
        const int hh  = ph + tap / 3 - 1;                                     \
        const int ww  = pw + tap % 3 - 1;                                     \
        const bool ok = (hh >= 0) & (hh < HH) & (ww >= 0) & (ww < WW);        \
        const int  li = ok ? lin[ibase + hh * WW + ww] : 0;                   \
        const __bf16* ga = ok ? (inB + ((ibase + (size_t)li) << 8) + cb + acg)\
                              : zpad;                                         \
        __builtin_amdgcn_global_load_async_to_lds_b128(                       \
            (AS1 v4i*)(AS1 void*)(void*)ga,                                   \
            (AS3 v4i*)(AS3 void*)&At[buf][arow][acg], 0, 0);                  \
        __builtin_amdgcn_global_load_async_to_lds_b128(                       \
            (AS1 v4i*)(AS1 void*)(void*)(wrow + k_),                          \
            (AS3 v4i*)(AS3 void*)&Bt[buf][bfr][bkg], 0, 0);                   \
        __builtin_amdgcn_global_load_async_to_lds_b128(                       \
            (AS1 v4i*)(AS1 void*)(void*)(wrow + k_ + 8),                      \
            (AS3 v4i*)(AS3 void*)&Bt[buf][bfr][bkg + 8], 0, 0);               \
    } while (0)

    ISSUE_STAGE(0, 0);
    int bufC = 0, bufL = 1;
    for (int kt = 0; kt < NKT; ++kt) {
        if (kt + 1 < NKT) {
            ISSUE_STAGE(bufL, kt + 1);
            __builtin_amdgcn_s_wait_asynccnt(3);   // stage kt has landed
        } else {
            __builtin_amdgcn_s_wait_asynccnt(0);
        }
        __syncthreads();                            // all waves' stage kt landed
#else
    for (int kt = 0; kt < NKT; ++kt) {
        const int bufC = 0;
        {   // synchronous staging
            const int k_  = kt * TILE_K;
            const int tap = k_ >> 8;
            const int cb  = k_ & (CC - 1);
            const int hh  = ph + tap / 3 - 1;
            const int ww  = pw + tap % 3 - 1;
            uint4 av = {0u, 0u, 0u, 0u};
            if (hh >= 0 && hh < HH && ww >= 0 && ww < WW) {
                const int li = lin[ibase + hh * WW + ww];
                av = *(const uint4*)(inB + ((ibase + (size_t)li) << 8) + cb + acg);
            }
            *(uint4*)&At[0][arow][acg] = av;
            *(uint4*)&Bt[0][bfr][bkg]     = *(const uint4*)(wrow + k_);
            *(uint4*)&Bt[0][bfr][bkg + 8] = *(const uint4*)(wrow + k_ + 8);
        }
        __syncthreads();
#endif
        // ---- fragments ----
        // A: lanes 0-15 hold K0-7,K16-23 ; lanes 16-31 hold K8-15,K24-31
        v8bf a0l = *(const v8bf*)&At[bufC][am0][half << 3];
        v8bf a0h = *(const v8bf*)&At[bufC][am0][16 + (half << 3)];
        v8bf a1l = *(const v8bf*)&At[bufC][am0 + 16][half << 3];
        v8bf a1h = *(const v8bf*)&At[bufC][am0 + 16][16 + (half << 3)];
        v16bf af0 = __builtin_shufflevector(a0l, a0h, 0,1,2,3,4,5,6,7,8,9,10,11,12,13,14,15);
        v16bf af1 = __builtin_shufflevector(a1l, a1h, 0,1,2,3,4,5,6,7,8,9,10,11,12,13,14,15);
        // B: lanes 0-15 hold K0-7,K16-23 ; lanes 16-31 hold K8-15,K24-31 (col = l16)
        v8bf b0l = *(const v8bf*)&Bt[bufC][bn0][half << 3];
        v8bf b0h = *(const v8bf*)&Bt[bufC][bn0][16 + (half << 3)];
        v8bf b1l = *(const v8bf*)&Bt[bufC][bn0 + 16][half << 3];
        v8bf b1h = *(const v8bf*)&Bt[bufC][bn0 + 16][16 + (half << 3)];
        v16bf bf0 = __builtin_shufflevector(b0l, b0h, 0,1,2,3,4,5,6,7,8,9,10,11,12,13,14,15);
        v16bf bf1 = __builtin_shufflevector(b1l, b1h, 0,1,2,3,4,5,6,7,8,9,10,11,12,13,14,15);

        acc00 = __builtin_amdgcn_wmma_f32_16x16x32_bf16(false, af0, false, bf0, (short)0, acc00, false, false);
        acc01 = __builtin_amdgcn_wmma_f32_16x16x32_bf16(false, af0, false, bf1, (short)0, acc01, false, false);
        acc10 = __builtin_amdgcn_wmma_f32_16x16x32_bf16(false, af1, false, bf0, (short)0, acc10, false, false);
        acc11 = __builtin_amdgcn_wmma_f32_16x16x32_bf16(false, af1, false, bf1, (short)0, acc11, false, false);

#if USE_ASYNC
        if (++bufC == 3) bufC = 0;
        if (++bufL == 3) bufL = 0;
#else
        __syncthreads();
#endif
    }

    // ---- epilogue: VGPR r holds row M = r + 8*half, col N = l16 ----
    const int f0 = nbase + (waveN << 5) + l16;
    const int f1 = f0 + 16;
    const float bb0 = bias[f0], bb1 = bias[f1];
    const int m0 = pbase + (waveM << 5) + (half << 3);
    #pragma unroll
    for (int r = 0; r < 8; ++r) {
        out[(size_t)(m0 + r) * FF + f0]      = acc00[r] + bb0;
        out[(size_t)(m0 + r) * FF + f1]      = acc01[r] + bb1;
        out[(size_t)(m0 + 16 + r) * FF + f0] = acc10[r] + bb0;
        out[(size_t)(m0 + 16 + r) * FF + f1] = acc11[r] + bb1;
    }
}

// -------------------------------------------------------------------------
extern "C" void kernel_launch(void* const* d_in, const int* in_sizes, int n_in,
                              void* d_out, int out_size, void* d_ws, size_t ws_size,
                              hipStream_t stream) {
    const float* inputs = (const float*)d_in[0];   // (16,128,128,256)
    const float* w_off  = (const float*)d_in[1];   // (3,3,256,2)
    const float* b_off  = (const float*)d_in[2];   // (2,)
    const float* w_def  = (const float*)d_in[3];   // (3,3,256,256)
    const float* b_def  = (const float*)d_in[4];   // (256,)
    float* out = (float*)d_out;                    // (16,128,128,256)

    // workspace: lin (1 MiB) | inputs bf16 (128 MiB) | wB bf16 | zero pad
    char* ws = (char*)d_ws;
    int*    lin  = (int*)ws;
    __bf16* inB  = (__bf16*)(ws + (size_t)(1 << 20));
    __bf16* wBp  = (__bf16*)(ws + (size_t)(1 << 20) + (size_t)NPIX * CC * 2);
    __bf16* zpad = wBp + (size_t)FF * KTOT;

    offsets_lin_kernel<<<NPIX / 256, 256, 0, stream>>>(inputs, w_off, b_off, lin);
    cvt_inputs_kernel<<<(size_t)NPIX * CC / (256 * 4), 256, 0, stream>>>(inputs, inB);
    cvt_wdef_kernel<<<FF * KTOT / 256, 256, 0, stream>>>(w_def, wBp, zpad);

    dim3 grid(NPIX / TILE_M, FF / TILE_N);
    dconv_gemm_kernel<<<grid, 256, 0, stream>>>(inB, wBp, b_def, lin, zpad, out);
}
